// ContrastiveLoss_6751688589321
// MI455X (gfx1250) — compile-verified, matrix operand order checked by reading, not run on hardware
//
#include <hip/hip_runtime.h>
#include <hip/hip_bf16.h>

// CDNA5 / gfx1250, wave32. WMMA f32<-f16 16x16x32 + async global->LDS staging.

typedef __attribute__((ext_vector_type(16))) _Float16 v16h;
typedef __attribute__((ext_vector_type(8)))  _Float16 v8h;
typedef __attribute__((ext_vector_type(8)))  float    v8f;
typedef __attribute__((ext_vector_type(4)))  int      v4i;

#define NROW 8192
#define CDIM 256
#define INV_T 10.0f      // 1 / TEMPERATURE
#define LDSROW 264       // padded LDS row stride (halves): 528B -> bank 4m%64, conflict-free

union Frag16 { v16h v; v8h h[2]; };

// --------- CDNA5 async global->LDS copy (ASYNCcnt path), 16B per lane ---------
__device__ __forceinline__ void async_g2l_b128(const _Float16* g, const _Float16* l) {
#if __has_builtin(__builtin_amdgcn_global_load_async_to_lds_b128)
  __builtin_amdgcn_global_load_async_to_lds_b128(
      (__attribute__((address_space(1))) v4i*)(uintptr_t)g,
      (__attribute__((address_space(3))) v4i*)(uintptr_t)(uint32_t)(uintptr_t)l,
      0, 0);
#else
  asm volatile("global_load_async_to_lds_b128 %0, %1, off"
               :: "v"((uint32_t)(uintptr_t)l), "v"((uint64_t)(uintptr_t)g)
               : "memory");
#endif
}
__device__ __forceinline__ void wait_async0() {
#if __has_builtin(__builtin_amdgcn_s_wait_asynccnt)
  __builtin_amdgcn_s_wait_asynccnt(0);
#else
  asm volatile("s_wait_asynccnt 0x0" ::: "memory");
#endif
}

// ---------------------------------------------------------------- zero scratch
__global__ __launch_bounds__(256) void zero_f32(float* __restrict__ p, int n) {
  int i = blockIdx.x * blockDim.x + threadIdx.x;
  if (i < n) p[i] = 0.0f;
}

// ------------------------------------------------- L2-normalize rows, emit f16
__global__ __launch_bounds__(256) void normalize_f16(const float* __restrict__ f,
                                                     _Float16* __restrict__ fn16) {
  const int row  = blockIdx.x * 8 + (threadIdx.x >> 5);
  const int lane = threadIdx.x & 31;
  const float4* rp = (const float4*)(f + (size_t)row * CDIM);
  float4 x0 = rp[lane * 2 + 0];
  float4 x1 = rp[lane * 2 + 1];
  float ss = x0.x*x0.x + x0.y*x0.y + x0.z*x0.z + x0.w*x0.w
           + x1.x*x1.x + x1.y*x1.y + x1.z*x1.z + x1.w*x1.w;
#pragma unroll
  for (int off = 16; off > 0; off >>= 1) ss += __shfl_xor(ss, off, 32);
  const float inv = 1.0f / fmaxf(sqrtf(ss), 1e-8f);
  v8h o;
  o[0] = (_Float16)(x0.x * inv); o[1] = (_Float16)(x0.y * inv);
  o[2] = (_Float16)(x0.z * inv); o[3] = (_Float16)(x0.w * inv);
  o[4] = (_Float16)(x1.x * inv); o[5] = (_Float16)(x1.y * inv);
  o[6] = (_Float16)(x1.z * inv); o[7] = (_Float16)(x1.w * inv);
  *(v8h*)(fn16 + (size_t)row * CDIM + lane * 8) = o;
}

// ---------------------------------------------------------------- main GEMM
// grid = (16, 128), block = 128 (4 waves). Block covers 4 row tiles (one per
// wave) x one 32-tile column strip. The shared B tile (16 cols x K=256, 8KB)
// is async-copied global->LDS once per block (4x less L2 read traffic than
// per-wave loads), double-buffered so the copy of tile t+1 overlaps compute
// on tile t. WMMA fragments: A from registers, B gathered from LDS
// (ds_load_b128, bank-conflict-free via 528B padded row stride).
__global__ __launch_bounds__(128) void supcon_tile(const _Float16* __restrict__ fn16,
                                                   const int* __restrict__ labels,
                                                   float* __restrict__ logits,
                                                   float* __restrict__ perfect,
                                                   float* __restrict__ S_exp,
                                                   float* __restrict__ S_pos,
                                                   float* __restrict__ S_cnt) {
  __shared__ _Float16 btile[2][16 * LDSROW];   // 2 x 8.25KB

  const int tid   = threadIdx.x;
  const int wave  = tid >> 5;
  const int lane  = tid & 31;
  const int m     = lane & 15;     // A row / B col within tile; also D col
  const int hi    = lane >> 4;     // half-wave selector (K sub-pattern)
  const int strip = blockIdx.x;                // column strip 0..15
  const int ib    = blockIdx.y * 4 + wave;     // row tile 0..511
  const int rbase = ib * 16 + 8 * hi;          // first of this lane's 8 out rows

  // cooperative-copy assignment: 16 rows x 512B; thread -> (row, 4 x 16B segs)
  const int crow = tid >> 3;       // 0..15
  const int cseg = tid & 7;        // 0..7 (segments cseg, cseg+8, +16, +24)

  // --- preload all A fragments for this row block (ISA 16-bit A layout:
  //     lane<16: halves = K0..7, K16..23 ; lane>=16: K8..15, K24..31)
  Frag16 afrag[8];
  {
    const _Float16* pa = fn16 + (size_t)(ib * 16 + m) * CDIM + hi * 8;
#pragma unroll
    for (int kk = 0; kk < 8; ++kk) {
      afrag[kk].h[0] = *(const v8h*)(pa + kk * 32);
      afrag[kk].h[1] = *(const v8h*)(pa + kk * 32 + 16);
    }
  }

  int rlab[8];
#pragma unroll
  for (int v = 0; v < 8; ++v) rlab[v] = labels[rbase + v];

  float sExp[8], sPos[8], sCnt[8];
#pragma unroll
  for (int v = 0; v < 8; ++v) { sExp[v] = 0.0f; sPos[v] = 0.0f; sCnt[v] = 0.0f; }

  // prologue: kick off async copy of tile 0 into buffer 0
  {
    const int jb0 = strip * 32;
    const _Float16* gsrc = fn16 + (size_t)(jb0 * 16 + crow) * CDIM + cseg * 8;
    const _Float16* ldst = &btile[0][crow * LDSROW + cseg * 8];
#pragma unroll
    for (int j = 0; j < 4; ++j) async_g2l_b128(gsrc + j * 64, ldst + j * 64);
  }

  for (int t = 0; t < 32; ++t) {
    const int p  = t & 1;
    const int jb = strip * 32 + t;

    wait_async0();      // this wave's portion of buffer p has landed in LDS
    __syncthreads();    // everyone's portion has landed; prev reads of buf p^1 done

    if (t < 31) {       // prefetch tile t+1 into the other buffer (overlaps compute)
      const _Float16* gsrc = fn16 + (size_t)((jb + 1) * 16 + crow) * CDIM + cseg * 8;
      const _Float16* ldst = &btile[p ^ 1][crow * LDSROW + cseg * 8];
#pragma unroll
      for (int j = 0; j < 4; ++j) async_g2l_b128(gsrc + j * 64, ldst + j * 64);
    }

    // --- 8 WMMA k-steps, B fragments gathered from LDS
    const _Float16* pb = &btile[p][m * LDSROW + hi * 8];
    v8f acc = {};
#pragma unroll
    for (int kk = 0; kk < 8; ++kk) {
      Frag16 b;
      b.h[0] = *(const v8h*)(pb + kk * 32);
      b.h[1] = *(const v8h*)(pb + kk * 32 + 16);
      acc = __builtin_amdgcn_wmma_f32_16x16x32_f16(
          /*neg_a=*/false, afrag[kk].v, /*neg_b=*/false, b.v,
          /*c_mod=*/(short)0, acc, /*reuse_a=*/false, /*reuse_b=*/false);
    }

    // --- epilogue: branchless, immediate-offset non-temporal stores
    const int   n    = jb * 16 + m;
    const int   clab = labels[n];
    float* __restrict__ pL = logits  + (size_t)rbase * NROW + n;
    float* __restrict__ pP = perfect + (size_t)rbase * NROW + n;
#pragma unroll
    for (int v = 0; v < 8; ++v) {
      const float lg   = acc[v] * INV_T;
      const float same = (rlab[v] == clab) ? 1.0f : 0.0f;
      const float offd = ((rbase + v) != n) ? 1.0f : 0.0f;
      __builtin_nontemporal_store(lg, pL + (size_t)v * NROW);
      __builtin_nontemporal_store(2.0f * same - 1.0f, pP + (size_t)v * NROW);
      const float pm = same * offd;      // pos_mask = same-class, off-diagonal
      sExp[v] += offd * __expf(lg);
      sPos[v] += pm * lg;
      sCnt[v] += pm;
    }
  }

  // reduce over the 16 lanes sharing each output row (xor<16 stays in half-wave)
#pragma unroll
  for (int off = 1; off < 16; off <<= 1) {
#pragma unroll
    for (int v = 0; v < 8; ++v) {
      sExp[v] += __shfl_xor(sExp[v], off, 32);
      sPos[v] += __shfl_xor(sPos[v], off, 32);
      sCnt[v] += __shfl_xor(sCnt[v], off, 32);
    }
  }
  if (m == 0) {
#pragma unroll
    for (int v = 0; v < 8; ++v) {
      atomicAdd(&S_exp[rbase + v], sExp[v]);
      atomicAdd(&S_pos[rbase + v], sPos[v]);
      atomicAdd(&S_cnt[rbase + v], sCnt[v]);
    }
  }
}

// ---------------------------------------------------------------- final loss
__global__ __launch_bounds__(256) void finalize_loss(const float* __restrict__ S_exp,
                                                     const float* __restrict__ S_pos,
                                                     const float* __restrict__ S_cnt,
                                                     float* __restrict__ out) {
  __shared__ float red[256];
  float s = 0.0f;
  for (int i = threadIdx.x; i < NROW; i += 256) {
    const float lden = __logf(S_exp[i]);
    s += (S_pos[i] - S_cnt[i] * lden) / (S_cnt[i] + 1e-6f);
  }
  red[threadIdx.x] = s;
  __syncthreads();
  for (int off = 128; off > 0; off >>= 1) {
    if ((int)threadIdx.x < off) red[threadIdx.x] += red[threadIdx.x + off];
    __syncthreads();
  }
  if (threadIdx.x == 0) out[0] = -red[0] / (float)NROW;  // T/baseT == 1
}

// ---------------------------------------------------------------- launcher
extern "C" void kernel_launch(void* const* d_in, const int* in_sizes, int n_in,
                              void* d_out, int out_size, void* d_ws, size_t ws_size,
                              hipStream_t stream) {
  (void)in_sizes; (void)n_in; (void)out_size; (void)ws_size;
  const float* feats  = (const float*)d_in[0];   // [8192, 1, 256] fp32
  const int*   labels = (const int*)d_in[1];     // [8192, 1] int32

  float* out     = (float*)d_out;
  float* loss    = out;                              // [1]
  float* logits  = out + 1;                          // [N, N]
  float* perfect = out + 1 + (size_t)NROW * NROW;    // [N, N]

  _Float16* fn16 = (_Float16*)d_ws;                            // 4 MB
  float* S_exp = (float*)((char*)d_ws + (size_t)NROW * CDIM * 2);
  float* S_pos = S_exp + NROW;
  float* S_cnt = S_pos + NROW;

  zero_f32<<<(3 * NROW + 255) / 256, 256, 0, stream>>>(S_exp, 3 * NROW);
  normalize_f16<<<NROW / 8, 256, 0, stream>>>(feats, fn16);
  supcon_tile<<<dim3(16, NROW / 64), 128, 0, stream>>>(fn16, labels, logits, perfect,
                                                       S_exp, S_pos, S_cnt);
  finalize_loss<<<1, 256, 0, stream>>>(S_exp, S_pos, S_cnt, loss);
}
